// Edge_Discriminator_1889785610729
// MI455X (gfx1250) — matrix operand order repaired and checked by reading, not compiled, not run on hardware
//
#include <hip/hip_runtime.h>
#include <hip/hip_bf16.h>

#define NN_DIM 4096
#define IN_DIM 512
#define HID    128
#define ALPHA_C 0.1f
#define BIAS_C  1e-4f
#define EOS_C   1e-10f

typedef _Float16 v8h  __attribute__((ext_vector_type(8)));
typedef _Float16 v16h __attribute__((ext_vector_type(16)));
typedef float    v8f  __attribute__((ext_vector_type(8)));
typedef float    v4f  __attribute__((ext_vector_type(4)));

// ---------------------------------------------------------------- utilities

__global__ void k_zero(float* __restrict__ p, int n) {
    int i = blockIdx.x * blockDim.x + threadIdx.x;
    if (i < n) p[i] = 0.0f;
}

// W1 [IN_DIM][HID] f32 -> W1t [HID][IN_DIM] f16 (transposed so WMMA B-tile
// loads become contiguous 16B vector loads)
__global__ void k_w1t(const float* __restrict__ W1, _Float16* __restrict__ W1t) {
    int i = blockIdx.x * blockDim.x + threadIdx.x;   // over IN_DIM*HID
    int k = i / HID, n = i % HID;
    W1t[n * IN_DIM + k] = (_Float16)W1[i];
}

// ------------------------------------------- emb = relu(F @ W1 + b1), f16 out
// One wave per 16x16 output tile. 2048 tiles -> 256 blocks x 8 waves.
__global__ void k_emb(const float* __restrict__ F,
                      const _Float16* __restrict__ W1t,
                      const float* __restrict__ b1,
                      _Float16* __restrict__ embH) {
    int wid  = threadIdx.x >> 5;
    int lane = threadIdx.x & 31;
    int wave = blockIdx.x * 8 + wid;
    int rowBase = (wave >> 3) * 16;        // 256 row tiles
    int colBase = (wave & 7) * 16;         // 8 col tiles (HID=128)
    int lr = lane & 15, hi = lane >> 4;
    int row = rowBase + lr;
    int col = colBase + lr;

    v8f c = {};
    for (int kb = 0; kb < IN_DIM; kb += 32) {
        // A layout (16-bit 16x32): lane holds row lr; K = {kb+hi*8+e, kb+16+hi*8+e}
        const float* ap0 = F + (size_t)row * IN_DIM + kb + hi * 8;
        const float* ap1 = ap0 + 16;
        v16h a, b;
#pragma unroll
        for (int e = 0; e < 8; ++e) { a[e] = (_Float16)ap0[e]; a[8 + e] = (_Float16)ap1[e]; }
        // B layout (16-bit 32x16): lane holds col lr; K = {kb+hi*8+e, kb+16+hi*8+e}
        const _Float16* bp = W1t + (size_t)col * IN_DIM + kb + hi * 8;
        v8h blo = *(const v8h*)bp, bhi = *(const v8h*)(bp + 16);
#pragma unroll
        for (int e = 0; e < 8; ++e) { b[e] = blo[e]; b[8 + e] = bhi[e]; }
        c = __builtin_amdgcn_wmma_f32_16x16x32_f16(false, a, false, b, (short)0, c, false, false);
    }
    float bias = b1[col];
#pragma unroll
    for (int r = 0; r < 8; ++r) {          // C layout: row = rowBase + hi*8 + r, col fixed
        float v = c[r] + bias;
        v = v > 0.0f ? v : 0.0f;
        embH[(size_t)(rowBase + hi * 8 + r) * HID + col] = (_Float16)v;
    }
}

// --------------------------- fused S = emb@embT + gumbel gating + row sums
// One wave handles a 16-row block x 256-column strip (16 tiles).
// 256 row blocks x 16 strips = 4096 waves -> 512 blocks x 8 waves.
__global__ void k_main(const _Float16* __restrict__ embH,
                       const float* __restrict__ adj,
                       const float* __restrict__ adj2,
                       const float* __restrict__ epsr,
                       float* __restrict__ wlp,
                       float* __restrict__ whp,
                       float* __restrict__ msk,
                       float* __restrict__ dlp,
                       float* __restrict__ dhp) {
    int wid  = threadIdx.x >> 5;
    int lane = threadIdx.x & 31;
    int wave = blockIdx.x * 8 + wid;
    int rowBase  = (wave >> 4) * 16;
    int colStrip = wave & 15;
    int lr = lane & 15, hi = lane >> 4;

    // Preload A = emb[rowBase..+15][0..127] in WMMA layout (4 K-blocks of 32)
    v16h A[4];
#pragma unroll
    for (int t = 0; t < 4; ++t) {
        const _Float16* base = embH + (size_t)(rowBase + lr) * HID + t * 32 + hi * 8;
        v8h lo = *(const v8h*)base;
        v8h hh = *(const v8h*)(base + 16);
#pragma unroll
        for (int e = 0; e < 8; ++e) { A[t][e] = lo[e]; A[t][8 + e] = hh[e]; }
    }

    float slp[8], shp[8];
#pragma unroll
    for (int r = 0; r < 8; ++r) { slp[r] = 0.0f; shp[r] = 0.0f; }

    for (int ct = 0; ct < 16; ++ct) {
        int colBase = (colStrip * 16 + ct) * 16;
        int col = colBase + lr;
        v8f c = {};
#pragma unroll
        for (int t = 0; t < 4; ++t) {
            // B = emb^T tile: lane holds column node `col`; K = {hi*8+e, 16+hi*8+e} along emb row
            const _Float16* bp = embH + (size_t)col * HID + t * 32 + hi * 8;
            v8h blo = *(const v8h*)bp, bhi = *(const v8h*)(bp + 16);
            v16h b;
#pragma unroll
            for (int e = 0; e < 8; ++e) { b[e] = blo[e]; b[8 + e] = bhi[e]; }
            c = __builtin_amdgcn_wmma_f32_16x16x32_f16(false, A[t], false, b, (short)0, c, false, false);
        }
        // Fused gumbel epilogue on the 16x16 S tile (C layout)
#pragma unroll
        for (int r = 0; r < 8; ++r) {
            int rowr = rowBase + hi * 8 + r;
            size_t off = (size_t)rowr * NN_DIM + col;
            float m1   = (adj[off]  != 0.0f) ? 1.0f : 0.0f;
            float m2   = (adj2[off] != 0.0f) ? 1.0f : 0.0f;
            float mult = m1 + m2;
            float em   = (mult > 0.0f) ? 1.0f : 0.0f;
            float r0   = epsr[off];
            float eps   = (2.0f * BIAS_C - 1.0f) * r0 + (1.0f - BIAS_C);
            float one_m = BIAS_C + (1.0f - 2.0f * BIAS_C) * r0;
            float gate  = __logf(eps) - __logf(one_m) + c[r] * mult;
            float sg    = 1.0f / (1.0f + __expf(-gate));
            float wl = sg * em;
            float wh = (1.0f - sg) * em;
            wlp[off] = wl; whp[off] = wh; msk[off] = em;
            slp[r] += wl; shp[r] += wh;
        }
    }
    // Row-sum reduce across the 16-lane group (xor masks 1..8 stay in-group)
#pragma unroll
    for (int r = 0; r < 8; ++r) {
        float a = slp[r], b = shp[r];
        for (int m = 8; m >= 1; m >>= 1) {
            a += __shfl_xor(a, m, 32);
            b += __shfl_xor(b, m, 32);
        }
        if (lr == 0) {
            atomicAdd(dlp + rowBase + hi * 8 + r, a);
            atomicAdd(dhp + rowBase + hi * 8 + r, b);
        }
    }
}

// dis = rsqrt(rowsum + 1 (identity diag) + EOS); d/dis arrays are contiguous
// [dlp|dhp] -> [dislp|dishp], so one launch covers both.
__global__ void k_dis(const float* __restrict__ d, float* __restrict__ dis, int n) {
    int i = blockIdx.x * blockDim.x + threadIdx.x;
    if (i < n) dis[i] = rsqrtf(d[i] + 1.0f + EOS_C);
}

// adj_lp = (w_lp+I)*dl_i*dl_j ; adj_hp = I - (w_hp+I)*dh_i*dh_j*mask*ALPHA
__global__ void k_final(const float* __restrict__ wlp,
                        const float* __restrict__ whp,
                        const float* __restrict__ msk,
                        const float* __restrict__ dislp,
                        const float* __restrict__ dishp,
                        float* __restrict__ alp,
                        float* __restrict__ ahp) {
    size_t idx = ((size_t)blockIdx.x * blockDim.x + threadIdx.x) * 4;
    int i = (int)(idx >> 12);            // / 4096
    int j = (int)(idx & (NN_DIM - 1));
    float di_l = dislp[i], di_h = dishp[i];
    v4f wl  = *(const v4f*)(wlp + idx);
    v4f wh  = *(const v4f*)(whp + idx);
    v4f em  = *(const v4f*)(msk + idx);
    v4f djl = *(const v4f*)(dislp + j);
    v4f djh = *(const v4f*)(dishp + j);
    v4f ol, oh;
#pragma unroll
    for (int t = 0; t < 4; ++t) {
        float I = (i == (j + t)) ? 1.0f : 0.0f;
        ol[t] = (wl[t] + I) * di_l * djl[t];
        oh[t] = I - (wh[t] + I) * di_h * djh[t] * em[t] * ALPHA_C;
    }
    *(v4f*)(alp + idx) = ol;
    *(v4f*)(ahp + idx) = oh;
}

extern "C" void kernel_launch(void* const* d_in, const int* in_sizes, int n_in,
                              void* d_out, int out_size, void* d_ws, size_t ws_size,
                              hipStream_t stream) {
    const float* F    = (const float*)d_in[0];
    const float* adj  = (const float*)d_in[1];
    const float* adj2 = (const float*)d_in[2];
    const float* epsr = (const float*)d_in[3];
    const float* W1   = (const float*)d_in[4];
    const float* b1   = (const float*)d_in[5];

    const size_t NN = (size_t)NN_DIM * NN_DIM;
    float* out = (float*)d_out;
    float* alp = out;
    float* ahp = out + NN;
    float* wlp = out + 2 * NN;
    float* whp = out + 3 * NN;
    float* msk = out + 4 * NN;

    char* ws = (char*)d_ws;
    _Float16* embH = (_Float16*)ws;  ws += (size_t)NN_DIM * HID * sizeof(_Float16);
    _Float16* W1t  = (_Float16*)ws;  ws += (size_t)HID * IN_DIM * sizeof(_Float16);
    float* dlp   = (float*)ws;  ws += NN_DIM * sizeof(float);
    float* dhp   = (float*)ws;  ws += NN_DIM * sizeof(float);   // contiguous after dlp
    float* dislp = (float*)ws;  ws += NN_DIM * sizeof(float);
    float* dishp = (float*)ws;  ws += NN_DIM * sizeof(float);   // contiguous after dislp
    (void)dhp; (void)dishp; (void)in_sizes; (void)n_in; (void)out_size; (void)ws_size;

    // 1. zero the two row-sum accumulators (contiguous 2N floats)
    k_zero<<<(2 * NN_DIM + 255) / 256, 256, 0, stream>>>(dlp, 2 * NN_DIM);
    // 2. transpose+convert W1
    k_w1t<<<(IN_DIM * HID) / 256, 256, 0, stream>>>(W1, W1t);
    // 3. emb = relu(F @ W1 + b1)  (WMMA, f16 out)
    k_emb<<<256, 256, 0, stream>>>(F, W1t, b1, embH);
    // 4. fused S = emb@embT + gumbel gating + row sums (WMMA)
    k_main<<<512, 256, 0, stream>>>(embH, adj, adj2, epsr, wlp, whp, msk, dlp, dhp);
    // 5. dis = rsqrt(d + 1 + eps) for both lp and hp (contiguous)
    k_dis<<<(2 * NN_DIM + 255) / 256, 256, 0, stream>>>(dlp, dislp, 2 * NN_DIM);
    // 6. normalized outputs
    k_final<<<(unsigned)(NN / 4 / 256), 256, 0, stream>>>(wlp, whp, msk, dislp, dishp, alp, ahp);
}
